// NLBlockND_34857954575064
// MI455X (gfx1250) — matrix operand order repaired and checked by reading, not compiled, not run on hardware
//
#include <hip/hip_runtime.h>
#include <hip/hip_bf16.h>

// ---------------------------------------------------------------------------
// NLBlockND (embedded-gaussian non-local block) for MI455X / gfx1250.
// GEMMs on v_wmma_f32_16x16x32_bf16; B-operands staged into LDS with the
// CDNA5 Tensor Data Mover (tensor_load_to_lds, TENSORcnt double-buffering).
// Attention is flash-style with the 16x4096 score strip in LDS (bf16),
// exploiting the 320 KB LDS per WGP.
// ---------------------------------------------------------------------------

#define N_   4
#define C_   512
#define CI_  256
#define L_   4096
#define EPS_ 1e-5f
#define SSTR 4104   // L_ + 8 (ushort elements), pad to dodge bank conflicts

typedef __attribute__((ext_vector_type(16))) __bf16 v16bf;
typedef __attribute__((ext_vector_type(8)))  __bf16 v8bf;
typedef __attribute__((ext_vector_type(8)))  float  v8f;
typedef __attribute__((ext_vector_type(4)))  unsigned int v4u;
typedef __attribute__((ext_vector_type(8)))  int v8i_;
typedef __attribute__((ext_vector_type(4)))  int v4i_;

__device__ __forceinline__ unsigned short f2bf(float f) {
  unsigned int u = __float_as_uint(f);
  u += 0x7FFFu + ((u >> 16) & 1u);          // round-to-nearest-even
  return (unsigned short)(u >> 16);
}
__device__ __forceinline__ float bf2f(unsigned short b) {
  return __uint_as_float(((unsigned int)b) << 16);
}

__device__ __forceinline__ v8f wmma_bf16(v16bf a, v16bf b, v8f c) {
  return __builtin_amdgcn_wmma_f32_16x16x32_bf16(
      /*neg_a=*/false, a, /*neg_b=*/false, b,
      /*c_mod=*/(short)0, c, /*reuse_a=*/false, /*reuse_b=*/false);
}

// A fragment (16x32 bf16, M x K). Lane l(<16): row M=l, K in {0..7,16..23};
// lane l+16: row M=l, K in {8..15,24..31}. rowk points at row start + k base.
__device__ __forceinline__ v16bf load_a_frag(const unsigned short* rowk, int lane) {
  const int s0 = (lane < 16) ? 0 : 8;
  v8bf lo = *(const v8bf*)(rowk + s0);
  v8bf hi = *(const v8bf*)(rowk + 16 + s0);
  return __builtin_shufflevector(lo, hi, 0, 1, 2, 3, 4, 5, 6, 7,
                                 8, 9, 10, 11, 12, 13, 14, 15);
}

__device__ __forceinline__ v16bf load_b_frag(const unsigned short* colk, int stride, int lane) {
  const int s0 = (lane < 16) ? 0 : 8;
  v16bf r;
  #pragma unroll
  for (int s = 0; s < 8; ++s) {
    r[s]     = __builtin_bit_cast(__bf16, colk[(s0 + s) * stride]);
    r[s + 8] = __builtin_bit_cast(__bf16, colk[(16 + s0 + s) * stride]);
  }
  return r;
}

__device__ __forceinline__ unsigned int rfl(unsigned int v) {
  return (unsigned int)__builtin_amdgcn_readfirstlane((int)v);
}

// ---------------------------------------------------------------------------
// Tensor Data Mover: 2D tile load Global -> LDS (bf16 elements, data_size=2B).
// Builds D# group0 (count/lds_addr/global_addr/type=2) and group1
// (data_size, tensor_dim0/1, tile_dim0/1, tensor_dim0_stride); groups 2/3
// zero (tile_dim2..4 unused -> 2D). Tracked by TENSORcnt.
// 6-arg builtin form (clang-23 / therock-10.0 toolchain).
// ---------------------------------------------------------------------------
__device__ __forceinline__ void tdm_load_2d(
    unsigned int lds_off,          // LDS byte offset of tile start
    const void* gaddr,             // global byte address of tile start
    unsigned int tile_d0,          // tile width  (elements)
    unsigned int tile_d1,          // tile height (rows)
    unsigned int tensor_d0,        // tensor width (elements)
    unsigned int tensor_d1,        // tensor height (rows)
    unsigned int stride0) {        // row stride (elements)
  const unsigned long long ga = (unsigned long long)gaddr;
  v4u g0;
  g0[0] = rfl(1u);                                   // count=1, user mode
  g0[1] = rfl(lds_off);                              // lds_addr
  g0[2] = rfl((unsigned int)ga);                     // global_addr[31:0]
  g0[3] = rfl(((unsigned int)(ga >> 32) & 0x01FFFFFFu) | 0x80000000u); // +type=2
  v8i_ g1;
  g1[0] = (int)rfl(1u << 16);                        // data_size=1 -> 2 bytes
  g1[1] = (int)rfl((tensor_d0 & 0xFFFFu) << 16);     // tensor_dim0[15:0]
  g1[2] = (int)rfl((tensor_d0 >> 16) | ((tensor_d1 & 0xFFFFu) << 16));
  g1[3] = (int)rfl((tensor_d1 >> 16) | (tile_d0 << 16));  // tile_dim0
  g1[4] = (int)rfl(tile_d1 & 0xFFFFu);               // tile_dim1 (tile_dim2=0)
  g1[5] = (int)rfl(stride0);                         // tensor_dim0_stride[31:0]
  g1[6] = 0;
  g1[7] = 0;
  v4i_ z4 = {0, 0, 0, 0};
  v8i_ z8 = {0, 0, 0, 0, 0, 0, 0, 0};
  __builtin_amdgcn_tensor_load_to_lds(g0, g1, z4, z4, z8, 0);
}

// ---------------------------------------------------------------------------
// 0) fp32 -> bf16 conversion of x and all weights, zero BN stats
// ---------------------------------------------------------------------------
__global__ __launch_bounds__(256) void prep_kernel(
    const float* __restrict__ x, const float* __restrict__ g_w,
    const float* __restrict__ th_w, const float* __restrict__ ph_w,
    const float* __restrict__ wz_w,
    unsigned short* __restrict__ xb, unsigned short* __restrict__ wb,
    unsigned short* __restrict__ wzb, float* __restrict__ stats) {
  const int i = blockIdx.x * 256 + threadIdx.x;
  if (i < N_ * C_ * L_) xb[i] = f2bf(x[i]);
  if (i < 3 * CI_ * C_) {                    // CI_*C_ = 2^17
    const int p = i >> 17;
    const int r = i & ((CI_ * C_) - 1);
    const float* w = (p == 0) ? g_w : (p == 1) ? th_w : ph_w;
    wb[i] = f2bf(w[r]);
  }
  if (i < C_ * CI_) wzb[i] = f2bf(wz_w[i]);
  if (i < 1024) stats[i] = 0.f;
}

// ---------------------------------------------------------------------------
// 1) projections: proj = W(CI x C) @ x(C x L) + b for g/theta/phi.
//    Block computes a 16(M) x 512(N) strip; TDM double-buffers the 32x512
//    bf16 B slab in LDS; 8 waves each own 64 N-columns.
//    g,theta written transposed (N,L,CI); phi written (N,CI,L).
// ---------------------------------------------------------------------------
#define PROJ_NK (C_ / 32)     // 16 K-steps
#define BSLAB   (32 * 512)    // ushorts per LDS slab (32 KB)

__global__ __launch_bounds__(256) void proj_kernel(
    const unsigned short* __restrict__ xb,   // (N, C, L)
    const unsigned short* __restrict__ wb,   // (3, CI, C)
    const float* __restrict__ g_b, const float* __restrict__ th_b,
    const float* __restrict__ ph_b,
    unsigned short* __restrict__ gT,         // (N, L, CI)
    unsigned short* __restrict__ tT,         // (N, L, CI)
    unsigned short* __restrict__ pb) {       // (N, CI, L)
  extern __shared__ char smem[];             // 2 x 32 KB B slabs
  unsigned short* bbuf = (unsigned short*)smem;

  const int lane = threadIdx.x & 31;
  const int w    = threadIdx.x >> 5;
  const int b    = blockIdx.x;
  const int nb   = b & 7;                    // 512-wide column block
  const int mt   = (b >> 3) & 15;
  const int n    = (b >> 7) & 3;
  const int p    = b >> 9;                   // 0=g 1=theta 2=phi
  const int cib  = mt * 16;
  const int lbb  = nb * 512;
  const int roff = (lane < 16) ? 0 : 8;

  const unsigned short* W = wb + p * (CI_ * C_);
  const unsigned short* X = xb + (size_t)n * (C_ * L_);
  const float* bias = (p == 0) ? g_b : (p == 1) ? th_b : ph_b;

  v8f acc[4];
  #pragma unroll
  for (int j = 0; j < 8; ++j) {
    const float bv = bias[cib + j + roff];
    #pragma unroll
    for (int t = 0; t < 4; ++t) acc[t][j] = bv;
  }

  const int m = cib + (lane & 15);

  if (w == 0)                                 // prologue: slab for k=0
    tdm_load_2d(0, X + lbb, 512, 32, L_, C_, L_);

  for (int kt = 0; kt < PROJ_NK; ++kt) {
    if (w == 0) {
      if (kt + 1 < PROJ_NK) {
        tdm_load_2d((unsigned int)(((kt + 1) & 1) * BSLAB * 2),
                    X + (size_t)(kt + 1) * 32 * L_ + lbb, 512, 32, L_, C_, L_);
        __builtin_amdgcn_s_wait_tensorcnt(1);  // current slab complete
      } else {
        __builtin_amdgcn_s_wait_tensorcnt(0);
      }
    }
    __syncthreads();                           // slab kt visible to all waves

    v16bf a = load_a_frag(W + m * C_ + kt * 32, lane);
    const unsigned short* bp = bbuf + (kt & 1) * BSLAB + w * 64 + (lane & 15);
    #pragma unroll
    for (int t = 0; t < 4; ++t) {
      v16bf bfrag = load_b_frag(bp + t * 16, 512, lane);
      acc[t] = wmma_bf16(a, bfrag, acc[t]);
    }
    __syncthreads();                           // done with slab before reuse
  }

  #pragma unroll
  for (int t = 0; t < 4; ++t) {
    #pragma unroll
    for (int j = 0; j < 8; ++j) {
      const int ci = cib + j + roff;                       // D row (M)
      const int l  = lbb + w * 64 + t * 16 + (lane & 15);  // D col (N)
      const unsigned short v = f2bf(acc[t][j]);
      if (p == 0)      gT[(size_t)n * (L_ * CI_) + (size_t)l * CI_ + ci] = v;
      else if (p == 1) tT[(size_t)n * (L_ * CI_) + (size_t)l * CI_ + ci] = v;
      else             pb[(size_t)n * (CI_ * L_) + (size_t)ci * L_ + l] = v;
    }
  }
}

// ---------------------------------------------------------------------------
// 2) attention, flash-style: one workgroup = one 16-row query tile.
//    Phase A: S(16 x 4096) = theta_tile^T @ phi, bf16 strip in LDS, row maxes.
//    Phase B: exp(s - max) in place, row sums.
//    Phase C: y = P @ g^T, scaled by 1/rowsum, scattered to yb (CI, L).
// ---------------------------------------------------------------------------
__global__ __launch_bounds__(256) void attn_kernel(
    const unsigned short* __restrict__ tT,   // (N, L, CI)
    const unsigned short* __restrict__ pb,   // (N, CI, L)
    const unsigned short* __restrict__ gT,   // (N, L, CI)
    unsigned short* __restrict__ yb) {       // (N, CI, L)
  extern __shared__ char smem[];
  unsigned short* S = (unsigned short*)smem;           // 16 * SSTR ushorts
  float* stage  = (float*)(smem + 16 * SSTR * 2);      // [8 waves][16 rows]
  float* rowmax = stage + 128;
  float* rowinv = rowmax + 16;

  const int lane = threadIdx.x & 31;
  const int w    = threadIdx.x >> 5;
  const int lt   = blockIdx.x & 255;
  const int n    = blockIdx.x >> 8;
  const int lb   = lt * 16;
  const int roff = (lane < 16) ? 0 : 8;

  const unsigned short* tTn = tT + (size_t)n * (L_ * CI_);
  const unsigned short* pbn = pb + (size_t)n * (CI_ * L_);
  const unsigned short* gTn = gT + (size_t)n * (L_ * CI_);
  unsigned short* ybn = yb + (size_t)n * (CI_ * L_);

  // ---- Phase A: scores for this wave's key range [w*512, w*512+512) ----
  v16bf afr[8];                               // theta^T tile, full K=256
  {
    const int l = lb + (lane & 15);
    #pragma unroll
    for (int kc = 0; kc < 8; ++kc)
      afr[kc] = load_a_frag(tTn + (size_t)l * CI_ + kc * 32, lane);
  }

  float rmax[8];
  #pragma unroll
  for (int j = 0; j < 8; ++j) rmax[j] = -3.4e38f;

  const int cb = w * 512;
  for (int mt2 = 0; mt2 < 32; ++mt2) {
    const int m0 = cb + mt2 * 16;
    v8f acc = {};
    #pragma unroll
    for (int kc = 0; kc < 8; ++kc) {
      v16bf b = load_b_frag(pbn + (size_t)(kc * 32) * L_ + m0 + (lane & 15), L_, lane);
      acc = wmma_bf16(afr[kc], b, acc);
    }
    const int col = m0 + (lane & 15);
    #pragma unroll
    for (int j = 0; j < 8; ++j) {
      const float v = acc[j];
      rmax[j] = fmaxf(rmax[j], v);
      S[(j + roff) * SSTR + col] = f2bf(v);
    }
  }
  #pragma unroll
  for (int j = 0; j < 8; ++j) {               // reduce over 16-lane half
    float v = rmax[j];
    v = fmaxf(v, __shfl_xor(v, 1, 32));
    v = fmaxf(v, __shfl_xor(v, 2, 32));
    v = fmaxf(v, __shfl_xor(v, 4, 32));
    v = fmaxf(v, __shfl_xor(v, 8, 32));
    rmax[j] = v;
  }
  if ((lane & 15) == 0) {
    #pragma unroll
    for (int j = 0; j < 8; ++j) stage[w * 16 + roff + j] = rmax[j];
  }
  __syncthreads();
  if (threadIdx.x < 16) {
    float v = -3.4e38f;
    for (int ww = 0; ww < 8; ++ww) v = fmaxf(v, stage[ww * 16 + threadIdx.x]);
    rowmax[threadIdx.x] = v;
  }
  __syncthreads();

  // ---- Phase B: softmax numerator + row sums over this wave's columns ----
  float rmx[16], rsum[16];
  #pragma unroll
  for (int r = 0; r < 16; ++r) { rmx[r] = rowmax[r]; rsum[r] = 0.f; }
  for (int kk = 0; kk < 16; ++kk) {
    const int col = cb + kk * 32 + lane;
    #pragma unroll
    for (int r = 0; r < 16; ++r) {
      const float s = bf2f(S[r * SSTR + col]);
      const float e = __expf(s - rmx[r]);
      S[r * SSTR + col] = f2bf(e);
      rsum[r] += e;
    }
  }
  #pragma unroll
  for (int r = 0; r < 16; ++r) {              // full 32-lane sum
    float v = rsum[r];
    v += __shfl_xor(v, 1, 32);
    v += __shfl_xor(v, 2, 32);
    v += __shfl_xor(v, 4, 32);
    v += __shfl_xor(v, 8, 32);
    v += __shfl_xor(v, 16, 32);
    rsum[r] = v;
  }
  if (lane == 0) {
    #pragma unroll
    for (int r = 0; r < 16; ++r) stage[w * 16 + r] = rsum[r];
  }
  __syncthreads();
  if (threadIdx.x < 16) {
    float v = 0.f;
    for (int ww = 0; ww < 8; ++ww) v += stage[ww * 16 + threadIdx.x];
    rowinv[threadIdx.x] = 1.0f / v;
  }
  __syncthreads();

  // ---- Phase C: y tile = P(16 x 4096) @ gT(4096 x 32) for ci range ----
  const int ci0 = w * 32;
  v8f acc0 = {}, acc1 = {};
  const unsigned short* Srow = S + (lane & 15) * SSTR;
  for (int k = 0; k < L_; k += 32) {
    v16bf a = load_a_frag(Srow + k, lane);    // ds_read_b128 x2
    const unsigned short* bp = gTn + (size_t)k * CI_ + ci0 + (lane & 15);
    v16bf b0 = load_b_frag(bp, CI_, lane);
    v16bf b1 = load_b_frag(bp + 16, CI_, lane);
    acc0 = wmma_bf16(a, b0, acc0);
    acc1 = wmma_bf16(a, b1, acc1);
  }
  #pragma unroll
  for (int j = 0; j < 8; ++j) {
    const int lrow = j + roff;
    const float inv = rowinv[lrow];
    const int l  = lb + lrow;
    const int c0 = ci0 + (lane & 15);
    ybn[(size_t)c0 * L_ + l]        = f2bf(acc0[j] * inv);
    ybn[(size_t)(c0 + 16) * L_ + l] = f2bf(acc1[j] * inv);
  }
}

// ---------------------------------------------------------------------------
// 3) w_y = wz(C x CI) @ y(CI x L) + b, fp32 out + BN sum/sumsq accumulation.
//    Same TDM double-buffered block structure as the projection GEMM.
// ---------------------------------------------------------------------------
#define WZ_NK (CI_ / 32)      // 8 K-steps

__global__ __launch_bounds__(256) void wz_kernel(
    const unsigned short* __restrict__ wzb,  // (C, CI)
    const unsigned short* __restrict__ yb,   // (N, CI, L)
    const float* __restrict__ wz_b,
    float* __restrict__ wy,                  // (N, C, L)
    float* __restrict__ stats) {             // sum[512] | sumsq[512]
  extern __shared__ char smem[];
  unsigned short* bbuf = (unsigned short*)smem;

  const int lane = threadIdx.x & 31;
  const int w    = threadIdx.x >> 5;
  const int b    = blockIdx.x;
  const int nb   = b & 7;
  const int mt   = (b >> 3) & 31;
  const int n    = b >> 8;
  const int cbase = mt * 16;
  const int lbb   = nb * 512;
  const int roff  = (lane < 16) ? 0 : 8;

  const unsigned short* Y = yb + (size_t)n * (CI_ * L_);
  float* WYn = wy + (size_t)n * (C_ * L_);

  v8f acc[4];
  #pragma unroll
  for (int j = 0; j < 8; ++j) {
    const float bv = wz_b[cbase + j + roff];
    #pragma unroll
    for (int t = 0; t < 4; ++t) acc[t][j] = bv;
  }

  const int m = cbase + (lane & 15);

  if (w == 0)
    tdm_load_2d(0, Y + lbb, 512, 32, L_, CI_, L_);

  for (int kt = 0; kt < WZ_NK; ++kt) {
    if (w == 0) {
      if (kt + 1 < WZ_NK) {
        tdm_load_2d((unsigned int)(((kt + 1) & 1) * BSLAB * 2),
                    Y + (size_t)(kt + 1) * 32 * L_ + lbb, 512, 32, L_, CI_, L_);
        __builtin_amdgcn_s_wait_tensorcnt(1);
      } else {
        __builtin_amdgcn_s_wait_tensorcnt(0);
      }
    }
    __syncthreads();

    v16bf a = load_a_frag(wzb + m * CI_ + kt * 32, lane);
    const unsigned short* bp = bbuf + (kt & 1) * BSLAB + w * 64 + (lane & 15);
    #pragma unroll
    for (int t = 0; t < 4; ++t) {
      v16bf bfrag = load_b_frag(bp + t * 16, 512, lane);
      acc[t] = wmma_bf16(a, bfrag, acc[t]);
    }
    __syncthreads();
  }

  float ssum[8], ssq[8];
  #pragma unroll
  for (int j = 0; j < 8; ++j) { ssum[j] = 0.f; ssq[j] = 0.f; }
  #pragma unroll
  for (int t = 0; t < 4; ++t) {
    #pragma unroll
    for (int j = 0; j < 8; ++j) {
      const float v = acc[t][j];
      WYn[(size_t)(cbase + j + roff) * L_ + lbb + w * 64 + t * 16 + (lane & 15)] = v;
      ssum[j] += v;
      ssq[j]  += v * v;
    }
  }
  #pragma unroll
  for (int j = 0; j < 8; ++j) {               // per-row reduce over 16 lanes
    float s = ssum[j], q = ssq[j];
    s += __shfl_xor(s, 1, 32);  q += __shfl_xor(q, 1, 32);
    s += __shfl_xor(s, 2, 32);  q += __shfl_xor(q, 2, 32);
    s += __shfl_xor(s, 4, 32);  q += __shfl_xor(q, 4, 32);
    s += __shfl_xor(s, 8, 32);  q += __shfl_xor(q, 8, 32);
    if ((lane & 15) == 0) {
      atomicAdd(&stats[cbase + j + roff], s);
      atomicAdd(&stats[C_ + cbase + j + roff], q);
    }
  }
}

// ---------------------------------------------------------------------------
// 4) batch-norm (training stats) + residual
// ---------------------------------------------------------------------------
__global__ __launch_bounds__(256) void bn_kernel(
    const float* __restrict__ wy, const float* __restrict__ stats,
    const float* __restrict__ gamma, const float* __restrict__ beta,
    const float* __restrict__ x, float* __restrict__ out) {
  const int i = blockIdx.x * 256 + threadIdx.x;
  if (i >= N_ * C_ * L_) return;
  const int c = (i >> 12) & (C_ - 1);         // L_ = 2^12
  const float cnt  = (float)(N_ * L_);
  const float mean = stats[c] / cnt;
  const float var  = stats[C_ + c] / cnt - mean * mean;
  const float v = (wy[i] - mean) * rsqrtf(var + EPS_);
  out[i] = v * gamma[c] + beta[c] + x[i];
}

// ---------------------------------------------------------------------------
extern "C" void kernel_launch(void* const* d_in, const int* in_sizes, int n_in,
                              void* d_out, int out_size, void* d_ws, size_t ws_size,
                              hipStream_t stream) {
  (void)in_sizes; (void)n_in; (void)out_size; (void)ws_size;
  const float* x     = (const float*)d_in[0];
  const float* g_w   = (const float*)d_in[1];
  const float* g_b   = (const float*)d_in[2];
  const float* th_w  = (const float*)d_in[3];
  const float* th_b  = (const float*)d_in[4];
  const float* ph_w  = (const float*)d_in[5];
  const float* ph_b  = (const float*)d_in[6];
  const float* wz_w  = (const float*)d_in[7];
  const float* wz_b  = (const float*)d_in[8];
  const float* gamma = (const float*)d_in[9];
  const float* beta  = (const float*)d_in[10];

  char* ws = (char*)d_ws;
  size_t off = 0;
  unsigned short* xb  = (unsigned short*)(ws + off); off += (size_t)N_ * C_ * L_ * 2;
  unsigned short* wb  = (unsigned short*)(ws + off); off += (size_t)3 * CI_ * C_ * 2;
  unsigned short* wzb = (unsigned short*)(ws + off); off += (size_t)C_ * CI_ * 2;
  unsigned short* gT  = (unsigned short*)(ws + off); off += (size_t)N_ * L_ * CI_ * 2;
  unsigned short* tT  = (unsigned short*)(ws + off); off += (size_t)N_ * L_ * CI_ * 2;
  unsigned short* pb  = (unsigned short*)(ws + off); off += (size_t)N_ * CI_ * L_ * 2;
  unsigned short* yb  = (unsigned short*)(ws + off); off += (size_t)N_ * CI_ * L_ * 2;
  float* wy    = (float*)(ws + off); off += (size_t)N_ * C_ * L_ * 4;
  float* stats = (float*)(ws + off); off += 1024 * 4;

  const int elems = N_ * C_ * L_;
  prep_kernel<<<(elems + 255) / 256, 256, 0, stream>>>(
      x, g_w, th_w, ph_w, wz_w, xb, wb, wzb, stats);

  const size_t gemm_smem = (size_t)2 * BSLAB * 2;   // 64 KB double buffer

  // 3 projections x 4 batches x 16 M-tiles x 8 N-blocks of 512 = 1536 blocks
  proj_kernel<<<1536, 256, gemm_smem, stream>>>(
      xb, wb, g_b, th_b, ph_b, gT, tT, pb);

  const size_t attn_smem = (size_t)16 * SSTR * 2 + (128 + 32) * sizeof(float);
  attn_kernel<<<N_ * (L_ / 16), 256, attn_smem, stream>>>(tT, pb, gT, yb);

  // 4 batches x 32 M-tiles x 8 N-blocks of 512 = 1024 blocks
  wz_kernel<<<1024, 256, gemm_smem, stream>>>(wzb, yb, wz_b, wy, stats);

  bn_kernel<<<(elems + 255) / 256, 256, 0, stream>>>(
      wy, stats, gamma, beta, x, (float*)d_out);
}
